// LI_8933531976414
// MI455X (gfx1250) — matrix operand order repaired and checked by reading, not compiled, not run on hardware
//
#include <hip/hip_runtime.h>

typedef __attribute__((ext_vector_type(4)))  float  v4f;
typedef __attribute__((ext_vector_type(8)))  float  v8f;
typedef __attribute__((ext_vector_type(8)))  __bf16 v8bf;
typedef __attribute__((ext_vector_type(16))) __bf16 v16bf;

#define B_   32
#define T_   1024
#define IN_  1024
#define OUT_ 1024
#define NT   64            // output columns per block
#define MC   64            // timesteps per chunk
#define KC   64            // K per LDS stage
#define NSTG (IN_ / KC)
#define LDB  (KC + 8)      // bf16 LDS row stride: 72 elems = 144 B (16B-aligned, bank-spread)
#define LDC  (NT + 4)

// Split 4 f32 into packed bf16 hi (RNE) / lo (trunc of exact residual) pairs.
// Error |x - hi - lo| <= 2^-16 |x|.  Pure 32-bit ALU + v_perm_b32 packing.
__device__ __forceinline__ void cvt_split(v4f v, uint2* hp, uint2* lp) {
    unsigned u0 = __float_as_uint(v[0]);
    unsigned u1 = __float_as_uint(v[1]);
    unsigned u2 = __float_as_uint(v[2]);
    unsigned u3 = __float_as_uint(v[3]);
    unsigned r0 = u0 + 0x7fffu + ((u0 >> 16) & 1u);   // RNE to bf16 in high half
    unsigned r1 = u1 + 0x7fffu + ((u1 >> 16) & 1u);
    unsigned r2 = u2 + 0x7fffu + ((u2 >> 16) & 1u);
    unsigned r3 = u3 + 0x7fffu + ((u3 >> 16) & 1u);
    float d0 = v[0] - __uint_as_float(r0 & 0xffff0000u);   // exact residual
    float d1 = v[1] - __uint_as_float(r1 & 0xffff0000u);
    float d2 = v[2] - __uint_as_float(r2 & 0xffff0000u);
    float d3 = v[3] - __uint_as_float(r3 & 0xffff0000u);
    hp->x = __builtin_amdgcn_perm(r1, r0, 0x07060302u);    // {r1.hi16, r0.hi16}
    hp->y = __builtin_amdgcn_perm(r3, r2, 0x07060302u);
    lp->x = __builtin_amdgcn_perm(__float_as_uint(d1), __float_as_uint(d0), 0x07060302u);
    lp->y = __builtin_amdgcn_perm(__float_as_uint(d3), __float_as_uint(d2), 0x07060302u);
}

__device__ __forceinline__ void cvt_store(unsigned short* __restrict__ ph,
                                          unsigned short* __restrict__ pl, v4f v) {
    uint2 h, l;
    cvt_split(v, &h, &l);
    *(uint2*)ph = h;
    *(uint2*)pl = l;
}

// A 16x32 bf16 fragment: lane(M=l%16,h=l/16): holds K=16h..16h+15 contiguous
__device__ __forceinline__ v16bf fragA(const unsigned short* base, int h) {
    v16bf r;
    *(v8bf*)&r         = *(const v8bf*)(base + 16 * h);
    *(((v8bf*)&r) + 1) = *(const v8bf*)(base + 16 * h + 8);
    return r;
}
// B 32x16 bf16 fragment (read as NxK = weight layout): lane holds K=16h..16h+15 contiguous
__device__ __forceinline__ v16bf fragB(const unsigned short* base, int h) {
    v16bf r;
    *(v8bf*)&r         = *(const v8bf*)(base + 16 * h);
    *(((v8bf*)&r) + 1) = *(const v8bf*)(base + 16 * h + 8);
    return r;
}

#define WMMA_BF16(A, Bm, C) \
    __builtin_amdgcn_wmma_f32_16x16x32_bf16(false, (A), false, (Bm), (short)0, (C), false, false)

// Pre-pass: split weight [OUT,IN] f32 into global bf16 hi/lo planes (4 MB total).
__global__ __launch_bounds__(256) void w_split_kernel(
    const float* __restrict__ w,
    unsigned short* __restrict__ wh,
    unsigned short* __restrict__ wl) {
    size_t idx = ((size_t)blockIdx.x * 256 + threadIdx.x) * 4;
    v4f v = *(const v4f*)(w + idx);
    uint2 h, l;
    cvt_split(v, &h, &l);
    *(uint2*)(wh + idx) = h;
    *(uint2*)(wl + idx) = l;
}

template <bool WPRE>
__global__ __launch_bounds__(256) void li_fused_kernel(
    const float* __restrict__ x,                 // [B,T,IN]
    const float* __restrict__ w,                 // [OUT,IN] (used when !WPRE)
    const unsigned short* __restrict__ wh_g,     // [OUT,IN] bf16 hi (WPRE)
    const unsigned short* __restrict__ wl_g,     // [OUT,IN] bf16 lo (WPRE)
    const float* __restrict__ bias,              // [OUT]
    const float* __restrict__ decay,             // [OUT]
    float* __restrict__ out)                     // [B,T,OUT]
{
    __shared__ __align__(16) unsigned short sXh[MC * LDB];
    __shared__ __align__(16) unsigned short sXl[MC * LDB];
    __shared__ __align__(16) unsigned short sWh[NT * LDB];
    __shared__ __align__(16) unsigned short sWl[NT * LDB];
    __shared__ __align__(16) float          sC[MC * LDC];

    const int tid  = threadIdx.x;
    const int lane = tid & 31;        // wave32
    const int wv   = tid >> 5;        // 0..7
    const int mblk = wv & 3;          // 16-row M sub-tile
    const int nblk = wv >> 2;         // 32-col N sub-tile
    const int l16  = lane & 15;
    const int hh   = lane >> 4;

    const int b   = blockIdx.y;
    const int n0g = blockIdx.x * NT;

    // scan state: thread tid<NT owns output column (n0g + tid)
    float u = 0.0f, dcy = 0.0f, omd = 0.0f, bi = 0.0f;
    if (tid < NT) {
        dcy = decay[n0g + tid];
        omd = 1.0f - dcy;
        bi  = bias[n0g + tid];
    }

    const int arow  = mblk * 16 + l16;
    const int brow0 = nblk * 32 + l16;
    // W staging map (WPRE): 64 rows x 4 groups of 16 bf16
    const int wr = tid >> 2;
    const int wc = (tid & 3) << 4;

    for (int tc = 0; tc < T_ / MC; ++tc) {
        v8f acc0 = {};
        v8f acc1 = {};
        const int t0 = tc * MC;

        for (int ks = 0; ks < NSTG; ++ks) {
            const int k0 = ks * KC;
            __syncthreads();   // prior readers of LDS planes done

            if (WPRE) {
                // copy pre-split W bf16 planes: no conversion VALU
                const size_t gw = (size_t)(n0g + wr) * IN_ + k0 + wc;
                uint4 a0 = *(const uint4*)(wh_g + gw);
                uint4 a1 = *(const uint4*)(wh_g + gw + 8);
                uint4 b0 = *(const uint4*)(wl_g + gw);
                uint4 b1 = *(const uint4*)(wl_g + gw + 8);
                *(uint4*)&sWh[wr * LDB + wc]     = a0;
                *(uint4*)&sWh[wr * LDB + wc + 8] = a1;
                *(uint4*)&sWl[wr * LDB + wc]     = b0;
                *(uint4*)&sWl[wr * LDB + wc + 8] = b1;
                if (ks + 1 < NSTG) {
                    __builtin_prefetch(wh_g + gw + KC, 0, 3);
                    __builtin_prefetch(wl_g + gw + KC, 0, 3);
                }
            }
            // stage 64x64 f32 X chunk, split into bf16 hi/lo planes
            #pragma unroll
            for (int i = 0; i < 4; ++i) {
                int q = tid + 256 * i;
                int r = q >> 4;
                int c = (q & 15) << 2;
                const float* xp = x + ((size_t)(b * T_ + t0 + r) * IN_ + k0 + c);
                v4f xv = *(const v4f*)xp;
                if (ks + 1 < NSTG) __builtin_prefetch(xp + KC, 0, 3);
                cvt_store(sXh + r * LDB + c, sXl + r * LDB + c, xv);
                if (!WPRE) {
                    const float* wp = w + ((size_t)(n0g + r) * IN_ + k0 + c);
                    v4f wv4 = *(const v4f*)wp;
                    cvt_store(sWh + r * LDB + c, sWl + r * LDB + c, wv4);
                }
            }
            __syncthreads();

            #pragma unroll
            for (int kb = 0; kb < KC; kb += 32) {
                v16bf Ah  = fragA(sXh + arow * LDB + kb, hh);
                v16bf Al  = fragA(sXl + arow * LDB + kb, hh);
                v16bf Bh0 = fragB(sWh + brow0 * LDB + kb, hh);
                v16bf Bl0 = fragB(sWl + brow0 * LDB + kb, hh);
                v16bf Bh1 = fragB(sWh + (brow0 + 16) * LDB + kb, hh);
                v16bf Bl1 = fragB(sWl + (brow0 + 16) * LDB + kb, hh);
                // x*w ~= xh*wh + xl*wh + xh*wl  (f32 accumulate)
                acc0 = WMMA_BF16(Ah, Bh0, acc0);
                acc1 = WMMA_BF16(Ah, Bh1, acc1);
                acc0 = WMMA_BF16(Al, Bh0, acc0);
                acc1 = WMMA_BF16(Al, Bh1, acc1);
                acc0 = WMMA_BF16(Ah, Bl0, acc0);
                acc1 = WMMA_BF16(Ah, Bl1, acc1);
            }
        }

        __syncthreads();
        // spill accumulators: C/D layout -> VGPR r holds M=r (lanes 0-15) / M=r+8
        #pragma unroll
        for (int r = 0; r < 8; ++r) {
            sC[(mblk * 16 + r + 8 * hh) * LDC + nblk * 32 + l16]      = acc0[r];
            sC[(mblk * 16 + r + 8 * hh) * LDC + nblk * 32 + 16 + l16] = acc1[r];
        }
        __syncthreads();

        // fused leaky-integrator scan over this chunk's 64 timesteps
        if (tid < NT) {
            float uu = u;
            float* op = out + ((size_t)(b * T_ + t0) * OUT_ + n0g + tid);
            #pragma unroll 4
            for (int t = 0; t < MC; ++t) {
                float cur = sC[t * LDC + tid] + bi;
                uu = dcy * uu + omd * cur;
                op[(size_t)t * OUT_] = uu;
            }
            u = uu;
        }
        // next chunk's first __syncthreads() protects LDS reuse
    }
}

extern "C" void kernel_launch(void* const* d_in, const int* in_sizes, int n_in,
                              void* d_out, int out_size, void* d_ws, size_t ws_size,
                              hipStream_t stream) {
    const float* x     = (const float*)d_in[0];
    const float* w     = (const float*)d_in[1];
    const float* bias  = (const float*)d_in[2];
    const float* decay = (const float*)d_in[3];
    float* out = (float*)d_out;

    unsigned short* wh = (unsigned short*)d_ws;
    unsigned short* wl = wh + (size_t)OUT_ * IN_;
    const size_t need = (size_t)OUT_ * IN_ * 2 * sizeof(unsigned short);

    dim3 grid(OUT_ / NT, B_);   // 16 x 32 = 512 blocks
    if (ws_size >= need) {
        w_split_kernel<<<(OUT_ * IN_) / 1024, 256, 0, stream>>>(w, wh, wl);
        li_fused_kernel<true><<<grid, 256, 0, stream>>>(x, w, wh, wl, bias, decay, out);
    } else {
        li_fused_kernel<false><<<grid, 256, 0, stream>>>(x, w, wh, wl, bias, decay, out);
    }
}
